// od_56040733278851
// MI455X (gfx1250) — matrix operand order repaired and checked by reading, not compiled, not run on hardware
//
#include <hip/hip_runtime.h>

typedef __attribute__((ext_vector_type(16))) _Float16 v16h;
typedef __attribute__((ext_vector_type(4)))  _Float16 v4h;
typedef __attribute__((ext_vector_type(8)))  float    v8f;
typedef __attribute__((ext_vector_type(4)))  float    v4f;

#define WMMA_F16(a, b, c) \
    __builtin_amdgcn_wmma_f32_16x16x32_f16(false, (a), false, (b), (short)0, (c), false, false)

// ---------------- utility kernels ----------------

__global__ void zero_f32(float* p, int n) {
    int t = blockIdx.x * blockDim.x + threadIdx.x;
    if (t < n) p[t] = 0.f;
}

// convert row-major f32 [K, Nin] -> f16 [K, Nout], zero-padding columns Nin..Nout
__global__ void cvt_w(const float* __restrict__ W, _Float16* __restrict__ Wh,
                      int K, int Nin, int Nout) {
    int t = blockIdx.x * blockDim.x + threadIdx.x;
    if (t >= K * Nout) return;
    int k = t / Nout, n = t % Nout;
    Wh[t] = (n < Nin) ? (_Float16)W[k * Nin + n] : (_Float16)0.f;
}

__global__ void pad_bias(const float* __restrict__ b, float* __restrict__ bp,
                         int Nin, int Nout) {
    int t = blockIdx.x * blockDim.x + threadIdx.x;
    if (t < Nout) bp[t] = (t < Nin) ? b[t] : 0.f;
}

// segment-sum over edges: out[dst[e]][f] += H[src[e]][f], F = 1<<fshift
__global__ void agg_edges(const float* __restrict__ H, const int* __restrict__ src,
                          const int* __restrict__ dst, float* __restrict__ out,
                          int E, int fshift) {
    int t = blockIdx.x * blockDim.x + threadIdx.x;
    int F = 1 << fshift;
    if (t >= (E << fshift)) return;
    int e = t >> fshift;
    int f = t & (F - 1);
    atomicAdd(&out[(size_t)dst[e] * F + f], H[(size_t)src[e] * F + f]);
}

// ---------------- WMMA GEMM: C = act(A[f32, MxK] * B[f16, KxN] + bias) ----------------
// one 16x16 output tile per wave; K % 32 == 0, M % 16 == 0, Ncols % 16 == 0
__global__ __launch_bounds__(256) void gemm_bias_act(
    const float* __restrict__ A, const _Float16* __restrict__ B,
    const float* __restrict__ bias, float* __restrict__ C,
    int M, int K, int Ncols, int relu) {
    int lane  = threadIdx.x & 31;
    int wave  = threadIdx.x >> 5;
    int ntN   = Ncols >> 4;
    int ntile = (M >> 4) * ntN;
    int tile  = blockIdx.x * 8 + wave;
    if (tile >= ntile) return;
    int mt = tile / ntN, nt = tile % ntN;

    int m     = lane & 15;             // A row within tile
    int nn    = lane & 15;             // B col / C col
    int mbase = (lane & 16) ? 8 : 0;   // C row split across lane halves
    int abase = (lane & 16) ? 8 : 0;   // A: K offset of lane half
    int bbase = (lane & 16) ? 8 : 0;   // B: K offset of lane half

    v8f acc = {};
    for (int kk = 0; kk < K; kk += 32) {
        v16h a, b;
        const float* Arow = A + (size_t)(mt * 16 + m) * K + kk + abase;
        #pragma unroll
        for (int t = 0; t < 8; t++) {
            a[t]     = (_Float16)Arow[t];        // K = abase + 0..7
            a[8 + t] = (_Float16)Arow[16 + t];   // K = abase + 16..23
        }
        const _Float16* Bp = B + (size_t)(kk + bbase) * Ncols + nt * 16 + nn;
        #pragma unroll
        for (int t = 0; t < 16; t++) b[t] = Bp[(size_t)(t < 8 ? t : t + 8) * Ncols];
        acc = WMMA_F16(a, b, acc);
    }

    float bb = bias[nt * 16 + nn];
    #pragma unroll
    for (int r = 0; r < 8; r++) {
        float v = acc[r] + bb;
        if (relu) v = fmaxf(v, 0.f);
        C[(size_t)(mt * 16 + mbase + r) * Ncols + nt * 16 + nn] = v;
    }
}

// ---------------- fused pairwise MLP ----------------
// 9000 waves total; each wave processes TPW=10 consecutive 16-pair tiles.
// Weight B-fragments (W1: 4 x v16h, W2: 8 x v16h) are hoisted into VGPRs once
// per wave; per tile the only LDS traffic is the X-gather and the 16x64 H tile.
// All inter-stage LDS deps are intra-wave (per-wave slices + in-order DS ops),
// so no block barrier is needed inside the tile loop.
#define MLP_TPW 10
__global__ __launch_bounds__(256) void mlp_fused(
    const float* __restrict__ Yq,       // [1200, 16] f32, cols 10..15 are zero
    const float* __restrict__ costs,    // [NP]
    const int*   __restrict__ indices,  // [NP]
    const float* __restrict__ W1, const float* __restrict__ b1,
    const float* __restrict__ W2, const float* __restrict__ b2,
    const float* __restrict__ W3, const float* __restrict__ b3,
    float* __restrict__ out) {
    __shared__ __align__(16) _Float16 sW1[32 * 64];   // K padded 21->32
    __shared__ __align__(16) _Float16 sW2[64 * 64];
    __shared__ float sW3[64];
    __shared__ float sB1[64];
    __shared__ float sB2[64];
    __shared__ __align__(16) _Float16 sX[8][16 * 32];
    __shared__ __align__(16) _Float16 sH[8][16 * 64];

    int tid  = threadIdx.x;
    int lane = tid & 31;
    int wave = tid >> 5;

    // ---- cooperatively stage f16 weights into LDS (vectorized x4) ----
    for (int t = tid; t < 32 * 16; t += 256) {            // sW1: 2048 halves
        int k = (t * 4) >> 6;
        v4h h;
        if (k < 21) {
            v4f v = *(const v4f*)(W1 + t * 4);
            h = (v4h){(_Float16)v.x, (_Float16)v.y, (_Float16)v.z, (_Float16)v.w};
        } else {
            h = (v4h){};
        }
        *(v4h*)(sW1 + t * 4) = h;
    }
    for (int t = tid; t < 64 * 16; t += 256) {            // sW2: 4096 halves
        v4f v = *(const v4f*)(W2 + t * 4);
        *(v4h*)(sW2 + t * 4) =
            (v4h){(_Float16)v.x, (_Float16)v.y, (_Float16)v.z, (_Float16)v.w};
    }
    if (tid < 64) { sW3[tid] = W3[tid]; sB1[tid] = b1[tid]; sB2[tid] = b2[tid]; }
    __syncthreads();

    int m     = lane & 15;
    int nn    = lane & 15;
    int mbase = (lane & 16) ? 8 : 0;
    int abase = (lane & 16) ? 8 : 0;
    int bbase = (lane & 16) ? 8 : 0;

    // ---- hoist weight B-fragments and per-lane scalars into registers ----
    v16h bw1[4];
    #pragma unroll
    for (int nt = 0; nt < 4; nt++)
        #pragma unroll
        for (int t = 0; t < 16; t++)
            bw1[nt][t] = sW1[(bbase + (t < 8 ? t : t + 8)) * 64 + nt * 16 + nn];
    v16h bw2[4][2];
    #pragma unroll
    for (int nt = 0; nt < 4; nt++)
        #pragma unroll
        for (int kk = 0; kk < 2; kk++)
            #pragma unroll
            for (int t = 0; t < 16; t++)
                bw2[nt][kk][t] = sW2[(kk * 32 + bbase + (t < 8 ? t : t + 8)) * 64 + nt * 16 + nn];
    float lb1[4], lb2[4], lw3[4];
    #pragma unroll
    for (int nt = 0; nt < 4; nt++) {
        lb1[nt] = sB1[nt * 16 + nn];
        lb2[nt] = sB2[nt * 16 + nn];
        lw3[nt] = sW3[nt * 16 + nn];
    }
    float b3v = b3[0];

    int wg = blockIdx.x * 8 + wave;                  // 0..8999

    for (int it = 0; it < MLP_TPW; it++) {
        int tile = wg * MLP_TPW + it;
        if (it + 1 < MLP_TPW)                        // gfx1250 global_prefetch_b8
            __builtin_prefetch(indices + (size_t)(tile + 1) * 16, 0, 3);

        // ---- build X tile: lanes 0..15, one pair row each ----
        if (lane < 16) {
            int row = lane;
            int idx = indices[(size_t)tile * 16 + row];
            int i = idx / 1200;
            int j = idx % 1200;
            #pragma unroll
            for (int c = 0; c < 10; c++) {
                sX[wave][row * 32 + c]      = (_Float16)Yq[(size_t)i * 16 + c];
                sX[wave][row * 32 + 10 + c] = (_Float16)Yq[(size_t)j * 16 + c];
            }
            sX[wave][row * 32 + 20] = (_Float16)costs[idx];
            #pragma unroll
            for (int c = 21; c < 32; c++) sX[wave][row * 32 + c] = (_Float16)0.f;
        }

        // ---- H1 = relu(X @ W1 + b1) : K=32, 4 N-tiles ----
        v16h a;
        #pragma unroll
        for (int t = 0; t < 8; t++) {
            a[t]     = sX[wave][m * 32 + abase + t];
            a[8 + t] = sX[wave][m * 32 + abase + 16 + t];
        }
        #pragma unroll
        for (int nt = 0; nt < 4; nt++) {
            v8f acc = {};
            acc = WMMA_F16(a, bw1[nt], acc);
            #pragma unroll
            for (int r = 0; r < 8; r++) {
                float v = fmaxf(acc[r] + lb1[nt], 0.f);
                sH[wave][(mbase + r) * 64 + nt * 16 + nn] = (_Float16)v;
            }
        }

        // ---- H2 = relu(H1 @ W2 + b2) : K=64, 4 N-tiles ----
        v8f acc2[4];
        #pragma unroll
        for (int kk = 0; kk < 2; kk++) {
            v16h a2;
            int kb = kk * 32 + abase;
            #pragma unroll
            for (int t = 0; t < 8; t++) {
                a2[t]     = sH[wave][m * 64 + kb + t];
                a2[8 + t] = sH[wave][m * 64 + kb + 16 + t];
            }
            #pragma unroll
            for (int nt = 0; nt < 4; nt++) {
                if (kk == 0) acc2[nt] = (v8f){};
                acc2[nt] = WMMA_F16(a2, bw2[nt][kk], acc2[nt]);
            }
        }

        // ---- out = relu(H2)·w3 + b3 : partial dot + wave32 butterfly reduce ----
        float part[8];
        #pragma unroll
        for (int r = 0; r < 8; r++) part[r] = 0.f;
        #pragma unroll
        for (int nt = 0; nt < 4; nt++) {
            #pragma unroll
            for (int r = 0; r < 8; r++) {
                float v = fmaxf(acc2[nt][r] + lb2[nt], 0.f);
                part[r] += v * lw3[nt];
            }
        }
        #pragma unroll
        for (int mask = 8; mask >= 1; mask >>= 1) {
            #pragma unroll
            for (int r = 0; r < 8; r++) part[r] += __shfl_xor(part[r], mask, 32);
        }
        if (nn == 0) {
            #pragma unroll
            for (int r = 0; r < 8; r++)
                out[(size_t)tile * 16 + mbase + r] = part[r] + b3v;
        }
    }
}

// ---------------- launcher ----------------

extern "C" void kernel_launch(void* const* d_in, const int* in_sizes, int n_in,
                              void* d_out, int out_size, void* d_ws, size_t ws_size,
                              hipStream_t stream) {
    const float* pois   = (const float*)d_in[0];
    const float* costs  = (const float*)d_in[1];
    const float* W_in   = (const float*)d_in[2];
    const float* b_in   = (const float*)d_in[3];
    const float* W_h    = (const float*)d_in[4];
    const float* b_h    = (const float*)d_in[5];
    const float* W_out  = (const float*)d_in[6];
    const float* b_out  = (const float*)d_in[7];
    const float* W1     = (const float*)d_in[8];
    const float* b1     = (const float*)d_in[9];
    const float* W2     = (const float*)d_in[10];
    const float* b2     = (const float*)d_in[11];
    const float* W3     = (const float*)d_in[12];
    const float* b3     = (const float*)d_in[13];
    const int*   edges  = (const int*)d_in[14];
    const int*   indices= (const int*)d_in[15];
    float*       out    = (float*)d_out;

    const int Nn = 1200, E = 38400, NP = 1200 * 1200;

    // workspace layout (bytes)
    char* ws = (char*)d_ws;
    float*    bufA  = (float*)ws;                         // [1200,256] agg scratch
    float*    bufB  = (float*)(ws + 1228800);             // [1200,256] activation
    float*    Yq    = (float*)(ws + 2457600);             // [1200,16] final node feats
    _Float16* WinH  = (_Float16*)(ws + 2534400);          // [128,256] f16
    _Float16* WhH   = (_Float16*)(ws + 2534400 + 65536);  // [256,256] f16
    _Float16* WoutH = (_Float16*)(ws + 2534400 + 65536 + 131072);          // [256,16] f16
    float*    boutP = (float*)(ws + 2534400 + 65536 + 131072 + 8192);      // [16]

    // stage f16 weights
    cvt_w<<<(128 * 256 + 255) / 256, 256, 0, stream>>>(W_in, WinH, 128, 256, 256);
    cvt_w<<<(256 * 256 + 255) / 256, 256, 0, stream>>>(W_h, WhH, 256, 256, 256);
    cvt_w<<<(256 * 16 + 255) / 256, 256, 0, stream>>>(W_out, WoutH, 256, 10, 16);
    pad_bias<<<1, 32, 0, stream>>>(b_out, boutP, 10, 16);

    const int* src = edges;
    const int* dst = edges + E;

    // layer 0: agg(pois) @ W_in
    zero_f32<<<(Nn * 128 + 255) / 256, 256, 0, stream>>>(bufA, Nn * 128);
    agg_edges<<<(E * 128) / 256, 256, 0, stream>>>(pois, src, dst, bufA, E, 7);
    gemm_bias_act<<<(75 * 16) / 8, 256, 0, stream>>>(bufA, WinH, b_in, bufB, 1200, 128, 256, 1);

    // layers 1,2: agg(y) @ W_h (shared weights)
    for (int l = 0; l < 2; l++) {
        zero_f32<<<(Nn * 256 + 255) / 256, 256, 0, stream>>>(bufA, Nn * 256);
        agg_edges<<<(E * 256) / 256, 256, 0, stream>>>(bufB, src, dst, bufA, E, 8);
        gemm_bias_act<<<(75 * 16) / 8, 256, 0, stream>>>(bufA, WhH, b_h, bufB, 1200, 256, 256, 1);
    }

    // layer 3: agg(y) @ W_out (no relu), padded to 16 cols
    zero_f32<<<(Nn * 256 + 255) / 256, 256, 0, stream>>>(bufA, Nn * 256);
    agg_edges<<<(E * 256) / 256, 256, 0, stream>>>(bufB, src, dst, bufA, E, 8);
    gemm_bias_act<<<(75 * 1 + 7) / 8, 256, 0, stream>>>(bufA, WoutH, boutP, Yq, 1200, 256, 16, 0);

    // fused pairwise MLP: 90000 tiles, 9000 waves x 10 tiles, 1125 blocks
    mlp_fused<<<(NP / 16) / (8 * MLP_TPW), 256, 0, stream>>>(Yq, costs, indices,
                                                             W1, b1, W2, b2, W3, b3, out);
}
